// LSTM_TEST_4337916969452
// MI455X (gfx1250) — matrix operand order repaired and checked by reading, not compiled, not run on hardware
//
#include <hip/hip_runtime.h>
#include <hip/hip_bf16.h>
#include <stdint.h>

// ---------------------------------------------------------------------------
// LSTM: B=128, T=1024, D=512, H=256, O=128, gate order (i,f,g,o)
// Phase 0: X, W_ih, W_hh -> f16 copies in workspace
// Phase 1: xg[T][B][4H] (f16) = X @ W_ih^T   (WMMA f16 GEMM, async-LDS DMA,
//                                             double-buffered, coalesced b128 stores)
// Phase 2: per-16-batch-tile private recurrence (WMMA f16 + f32 cell state)
// Phase 3: out[B][O] = h_T @ W_out^T + b_out
// ---------------------------------------------------------------------------

typedef __attribute__((ext_vector_type(16))) _Float16 v16h;
typedef __attribute__((ext_vector_type(8)))  _Float16 v8h;
typedef __attribute__((ext_vector_type(8)))  float    v8f;

#define LSTM_B   128
#define LSTM_T   1024
#define LSTM_D   512
#define LSTM_H   256
#define LSTM_NG  1024   // 4*H
#define LSTM_O   128

static __device__ __forceinline__ v16h join16(v8h lo, v8h hi) {
    v16h r;
#pragma unroll
    for (int i = 0; i < 8; ++i) { r[i] = lo[i]; r[i + 8] = hi[i]; }
    return r;
}

static __device__ __forceinline__ v8f wmma_f16(v16h a, v16h b, v8f c) {
    return __builtin_amdgcn_wmma_f32_16x16x32_f16(
        /*neg_a=*/false, a, /*neg_b=*/false, b,
        /*c_mod=*/(short)0, c, /*reuse_a=*/false, /*reuse_b=*/false);
}

static __device__ __forceinline__ v8f zero8f() {
    v8f z;
#pragma unroll
    for (int i = 0; i < 8; ++i) z[i] = 0.0f;
    return z;
}

static __device__ __forceinline__ float fast_sigmoid(float x) {
    return __builtin_amdgcn_rcpf(1.0f + __expf(-x));
}
static __device__ __forceinline__ float fast_tanh(float x) {
    return 1.0f - 2.0f * __builtin_amdgcn_rcpf(1.0f + __expf(2.0f * x));
}

// ---- CDNA5 async global->LDS DMA (ASYNCcnt path), 16B per lane ------------
static __device__ __forceinline__ void async_b128(uint32_t lds_off,
                                                  const _Float16* gaddr) {
    asm volatile("global_load_async_to_lds_b128 %0, %1, off"
                 :: "v"(lds_off), "v"(gaddr)
                 : "memory");
}
#define WAIT_ASYNC(n) asm volatile("s_wait_asynccnt " #n ::: "memory")

static __device__ __forceinline__ uint32_t lds_addr32(const void* p) {
    return (uint32_t)(uintptr_t)p;   // LDS aperture: addr[31:0] = LDS offset
}

// ---------------------------------------------------------------------------
// Kernel 0a: X (f32) -> Xh (f16), 8 elements per thread, vectorized.
// ---------------------------------------------------------------------------
__global__ __launch_bounds__(256) void cvt_x_kernel(const float* __restrict__ X,
                                                    _Float16* __restrict__ Xh) {
    const size_t i = ((size_t)blockIdx.x * 256 + threadIdx.x) * 8;
    float4 f0 = *(const float4*)(X + i);
    float4 f1 = *(const float4*)(X + i + 4);
    v8h h;
    h[0]=(_Float16)f0.x; h[1]=(_Float16)f0.y; h[2]=(_Float16)f0.z; h[3]=(_Float16)f0.w;
    h[4]=(_Float16)f1.x; h[5]=(_Float16)f1.y; h[6]=(_Float16)f1.z; h[7]=(_Float16)f1.w;
    *(v8h*)(Xh + i) = h;
}

// ---------------------------------------------------------------------------
// Kernel 0b: weights -> f16.
// ---------------------------------------------------------------------------
__global__ void cvt_weights_kernel(const float* __restrict__ Wih,
                                   const float* __restrict__ Whh,
                                   _Float16* __restrict__ WihH,
                                   _Float16* __restrict__ WhhH) {
    int i = blockIdx.x * blockDim.x + threadIdx.x;
    if (i < LSTM_NG * LSTM_D) WihH[i] = (_Float16)Wih[i];
    if (i < LSTM_NG * LSTM_H) WhhH[i] = (_Float16)Whh[i];
}

// ---------------------------------------------------------------------------
// Kernel 1: xg[t][b][n] = sum_k X[b][t][k] * W_ih[n][k]   (f16 output)
// Block: 256 threads (8 waves). Block tile: M=128 (all b for one t), N=64.
// K chunks of 32; double-buffered async global->LDS DMA; LDS-staged coalesced
// b128 stores.
// ---------------------------------------------------------------------------
__global__ __launch_bounds__(256) void xproj_kernel(
        const _Float16* __restrict__ Xh,
        const _Float16* __restrict__ WihH,
        _Float16* __restrict__ xg) {
    __shared__ _Float16 Abuf[2][128][40];  // [m][k], padded stride 40 halves
    __shared__ _Float16 Bbuf[2][64][40];   // [n][k]
    __shared__ _Float16 Cs[128][72];       // store staging, stride 72 halves

    const int t   = blockIdx.y;
    const int nb  = blockIdx.x;            // N block: cols [64*nb, 64*nb+64)
    const int tid = threadIdx.x;
    const int wave = tid >> 5;
    const int lane = tid & 31;
    const int ln = lane & 15, hi = lane >> 4;

    // Per-thread DMA chunk assignments (3 x 16B per thread per K-chunk):
    //   A tile: 128 rows x 32 halves = 512 chunks -> 2 per thread
    //   B tile:  64 rows x 32 halves = 256 chunks -> 1 per thread
    const int ar0 = (tid * 2) >> 2, aq0 = (tid * 2) & 3;        // chunk tid*2
    const int ar1 = (tid * 2 + 1) >> 2, aq1 = (tid * 2 + 1) & 3;
    const int bn = tid >> 2, bq = tid & 3;
    const _Float16* agp0 = Xh + ((size_t)ar0 * LSTM_T + t) * LSTM_D + aq0 * 8;
    const _Float16* agp1 = Xh + ((size_t)ar1 * LSTM_T + t) * LSTM_D + aq1 * 8;
    const _Float16* bgp  = WihH + ((size_t)(nb * 64 + bn)) * LSTM_D + bq * 8;

    v8f acc[4];
#pragma unroll
    for (int n = 0; n < 4; ++n) acc[n] = zero8f();

    // prologue: prefetch K-chunk 0 into buffer 0
    async_b128(lds_addr32(&Abuf[0][ar0][aq0 * 8]), agp0);
    async_b128(lds_addr32(&Abuf[0][ar1][aq1 * 8]), agp1);
    async_b128(lds_addr32(&Bbuf[0][bn][bq * 8]),  bgp);

    for (int it = 0; it < 16; ++it) {
        const int cur = it & 1;
        if (it + 1 < 16) {
            const int kc = (it + 1) * 32;
            async_b128(lds_addr32(&Abuf[cur ^ 1][ar0][aq0 * 8]), agp0 + kc);
            async_b128(lds_addr32(&Abuf[cur ^ 1][ar1][aq1 * 8]), agp1 + kc);
            async_b128(lds_addr32(&Bbuf[cur ^ 1][bn][bq * 8]),  bgp + kc);
            WAIT_ASYNC(3);               // chunk `it` DMA retired (this wave)
        } else {
            WAIT_ASYNC(0);
        }
        __syncthreads();                 // all waves' DMA for `cur` visible

        const int m0 = wave * 16;
        v16h a = join16(*(const v8h*)&Abuf[cur][m0 + ln][hi * 8],
                        *(const v8h*)&Abuf[cur][m0 + ln][16 + hi * 8]);
#pragma unroll
        for (int nt = 0; nt < 4; ++nt) {
            v16h b = join16(*(const v8h*)&Bbuf[cur][nt * 16 + ln][hi * 8],
                            *(const v8h*)&Bbuf[cur][nt * 16 + ln][16 + hi * 8]);
            acc[nt] = wmma_f16(a, b, acc[nt]);
        }
        __syncthreads();                 // done reading `cur`; may overwrite
    }

    // ---- stage C tile in LDS (f16), then fully coalesced b128 stores ------
#pragma unroll
    for (int nt = 0; nt < 4; ++nt) {
#pragma unroll
        for (int v = 0; v < 8; ++v) {
            Cs[wave * 16 + v + 8 * hi][nt * 16 + ln] = (_Float16)acc[nt][v];
        }
    }
    __syncthreads();
    // 128 rows x 64 halves = 1024 x 16B chunks; 4 per thread
#pragma unroll
    for (int u = 0; u < 4; ++u) {
        const int c = tid * 4 + u;
        const int r = c >> 3, q = c & 7;
        v8h val = *(const v8h*)&Cs[r][q * 8];
        *(v8h*)(xg + ((size_t)t * LSTM_B + r) * LSTM_NG + nb * 64 + q * 8) = val;
    }
}

// ---------------------------------------------------------------------------
// Kernel 2: private recurrence. 8 blocks x 1024 threads (32 waves).
// Block handles batch rows [16*blk, 16*blk+16) for all T steps — batches are
// independent so no inter-block sync exists. Wave w computes gate columns
// [32w, 32w+32); all threads then do the elementwise cell update with c in
// f32 registers.
// ---------------------------------------------------------------------------
__global__ __launch_bounds__(1024) void lstm_rec_kernel(
        const _Float16* __restrict__ xg,
        const _Float16* __restrict__ WhhH,
        const float* __restrict__ b_ih,
        const float* __restrict__ b_hh,
        float* __restrict__ hT) {
    __shared__ _Float16 hA[16][264];     // h in f16, [b][k], padded
    __shared__ float gates[16][1028];    // h @ Whh^T partials, padded

    const int tid  = threadIdx.x;
    const int wave = tid >> 5;
    const int lane = tid & 31;
    const int ln = lane & 15, hi = lane >> 4;
    const int B0 = blockIdx.x * 16;

    for (int i = tid; i < 16 * 264; i += 1024) ((_Float16*)hA)[i] = (_Float16)0.0f;

    const int jq = tid & 63;
    const int bb = tid >> 6;
    float c_state[4] = {0.f, 0.f, 0.f, 0.f};
    float bi[4], bf[4], bg[4], bo[4];
#pragma unroll
    for (int k = 0; k < 4; ++k) {
        const int j = jq + 64 * k;
        bi[k] = b_ih[j]       + b_hh[j];
        bf[k] = b_ih[256 + j] + b_hh[256 + j];
        bg[k] = b_ih[512 + j] + b_hh[512 + j];
        bo[k] = b_ih[768 + j] + b_hh[768 + j];
    }
    __syncthreads();

    for (int t = 0; t < LSTM_T; ++t) {
        // gates[0:16][32w:32w+32] = h @ Whh^T  (K = 256 = 8 WMMA K-tiles)
        v8f acc0 = zero8f(), acc1 = zero8f();
#pragma unroll
        for (int kt = 0; kt < 8; ++kt) {
            v16h a = join16(*(const v8h*)&hA[ln][kt * 32 + hi * 8],
                            *(const v8h*)&hA[ln][kt * 32 + 16 + hi * 8]);
            const _Float16* bp0 =
                WhhH + ((size_t)(wave * 32 + ln)) * LSTM_H + kt * 32 + hi * 8;
            const _Float16* bp1 = bp0 + 16 * LSTM_H;
            v16h b0 = join16(*(const v8h*)bp0, *(const v8h*)(bp0 + 16));
            v16h b1 = join16(*(const v8h*)bp1, *(const v8h*)(bp1 + 16));
            acc0 = wmma_f16(a, b0, acc0);
            acc1 = wmma_f16(a, b1, acc1);
        }
#pragma unroll
        for (int v = 0; v < 8; ++v) {
            const int m = v + 8 * hi;
            gates[m][wave * 32 + ln]      = acc0[v];
            gates[m][wave * 32 + 16 + ln] = acc1[v];
        }
        __syncthreads();

        const _Float16* xrow = xg + ((size_t)t * LSTM_B + B0 + bb) * LSTM_NG;
#pragma unroll
        for (int k = 0; k < 4; ++k) {
            const int j = jq + 64 * k;
            const float gi = gates[bb][j]       + (float)xrow[j]       + bi[k];
            const float gf = gates[bb][j + 256] + (float)xrow[j + 256] + bf[k];
            const float gg = gates[bb][j + 512] + (float)xrow[j + 512] + bg[k];
            const float go = gates[bb][j + 768] + (float)xrow[j + 768] + bo[k];
            const float iv = fast_sigmoid(gi);
            const float fv = fast_sigmoid(gf);
            const float gv = fast_tanh(gg);
            const float ov = fast_sigmoid(go);
            c_state[k] = fv * c_state[k] + iv * gv;
            const float hv = ov * fast_tanh(c_state[k]);
            hA[bb][j] = (_Float16)hv;
            if (t == LSTM_T - 1) hT[(size_t)(B0 + bb) * LSTM_H + j] = hv;
        }
        __syncthreads();
    }
}

// ---------------------------------------------------------------------------
// Kernel 3: out[b][o] = b_out[o] + sum_k hT[b][k] * W_out[o][k]
// ---------------------------------------------------------------------------
__global__ __launch_bounds__(256) void outproj_kernel(
        const float* __restrict__ hT,
        const float* __restrict__ Wout,
        const float* __restrict__ bout,
        float* __restrict__ out) {
    const int idx = blockIdx.x * 256 + threadIdx.x;
    const int b = idx >> 7, o = idx & 127;
    const float* hp = hT + (size_t)b * LSTM_H;
    const float* wp = Wout + (size_t)o * LSTM_H;
    float s = bout[o];
#pragma unroll 8
    for (int k = 0; k < LSTM_H; ++k) s = fmaf(hp[k], wp[k], s);
    out[idx] = s;
}

// ---------------------------------------------------------------------------
extern "C" void kernel_launch(void* const* d_in, const int* in_sizes, int n_in,
                              void* d_out, int out_size, void* d_ws, size_t ws_size,
                              hipStream_t stream) {
    const float* X     = (const float*)d_in[0];   // [B,T,D]
    const float* W_ih  = (const float*)d_in[1];   // [4H,D]
    const float* W_hh  = (const float*)d_in[2];   // [4H,H]
    const float* b_ih  = (const float*)d_in[3];   // [4H]
    const float* b_hh  = (const float*)d_in[4];   // [4H]
    const float* W_out = (const float*)d_in[5];   // [O,H]
    const float* b_out = (const float*)d_in[6];   // [O]
    float* out = (float*)d_out;                   // [B,O]

    // workspace carve-up
    char* w = (char*)d_ws;
    _Float16* xg   = (_Float16*)w;                               // 256 MB: [T][B][4H]
    size_t off = (size_t)LSTM_T * LSTM_B * LSTM_NG * sizeof(_Float16);
    _Float16* Xh   = (_Float16*)(w + off);                       // 128 MB: [B][T][D]
    off += (size_t)LSTM_B * LSTM_T * LSTM_D * sizeof(_Float16);
    _Float16* WihH = (_Float16*)(w + off);                       // 1 MB
    off += (size_t)LSTM_NG * LSTM_D * sizeof(_Float16);
    _Float16* WhhH = (_Float16*)(w + off);                       // 0.5 MB
    off += (size_t)LSTM_NG * LSTM_H * sizeof(_Float16);
    float* hT = (float*)(w + off);                               // 128 KB

    const size_t nX = (size_t)LSTM_B * LSTM_T * LSTM_D;          // 67,108,864
    cvt_x_kernel<<<(unsigned)(nX / (256 * 8)), 256, 0, stream>>>(X, Xh);

    cvt_weights_kernel<<<(LSTM_NG * LSTM_D) / 256, 256, 0, stream>>>(
        W_ih, W_hh, WihH, WhhH);

    xproj_kernel<<<dim3(LSTM_NG / 64, LSTM_T), 256, 0, stream>>>(Xh, WihH, xg);

    lstm_rec_kernel<<<LSTM_B / 16, 1024, 0, stream>>>(xg, WhhH, b_ih, b_hh, hT);

    outproj_kernel<<<(LSTM_B * LSTM_O) / 256, 256, 0, stream>>>(hT, W_out, b_out, out);
}
